// RelPosAttention_78512002171103
// MI455X (gfx1250) — compile-verified, matrix operand order checked by reading, not run on hardware
//
#include <hip/hip_runtime.h>
#include <stdint.h>

typedef __attribute__((ext_vector_type(16))) _Float16 v16h;
typedef __attribute__((ext_vector_type(8)))  float    v8f;

union F16Frag { v16h v; _Float16 h[16]; unsigned u[8]; };

__device__ __forceinline__ int kpair(int j, int hi) {
  // even K index held by VGPR j (halves 2j,2j+1) for lane half `hi`
  return ((j < 4) ? (2 * j) : (16 + 2 * (j - 4))) + 8 * hi;
}

__device__ __forceinline__ v8f v8f_zero() {
  v8f z = {0.f, 0.f, 0.f, 0.f, 0.f, 0.f, 0.f, 0.f};
  return z;
}

// CDNA5 async copy: global -> LDS, 16 bytes per lane, tracked by ASYNCcnt.
__device__ __forceinline__ void async_copy16(unsigned lds_off, const void* gptr) {
  asm volatile("global_load_async_to_lds_b128 %0, %1, off"
               :: "v"(lds_off), "v"(gptr) : "memory");
}

#define L_SEQ 197
#define L_PAD 208
#define L_VPAD 224
#define NH 12
#define HD 64
#define CDIM 768
#define NROWS (256 * 197)
#define NBH 3072

// ---------------------------------------------------------------- X -> f16
__global__ void xprep_kernel(const float* __restrict__ X, _Float16* __restrict__ Xh) {
  long i = ((long)blockIdx.x * 256 + threadIdx.x) * 4;   // NROWS*CDIM exact
  float4 v = *(const float4*)(X + i);
  Xh[i + 0] = (_Float16)v.x;
  Xh[i + 1] = (_Float16)v.y;
  Xh[i + 2] = (_Float16)v.z;
  Xh[i + 3] = (_Float16)v.w;
}

// ---------------------------------------------------------------- weight prep
// W is [in=768][out=768] fp32 (x @ W). Store Wt[out][in] f16.
__global__ void wprep_kernel(const float* __restrict__ W, _Float16* __restrict__ Wt) {
  int i = blockIdx.x * 256 + threadIdx.x;   // 768*768 exact
  int r = i / CDIM;                         // input dim
  int c = i - r * CDIM;                     // output dim
  Wt[(long)c * CDIM + r] = (_Float16)W[i];
}

// ---------------------------------------------------------------- bias build
__global__ void bias_kernel(const float* __restrict__ rel_table,
                            const float* __restrict__ c2p,
                            const float* __restrict__ p2c,
                            const float* __restrict__ c2c,
                            float* __restrict__ biasF) {
  int h = blockIdx.y;
  int i = blockIdx.x * 256 + threadIdx.x;
  if (i >= L_SEQ * L_SEQ) return;
  int q = i / L_SEQ, k = i - q * L_SEQ;
  float v;
  if (q == 0 && k == 0)      v = c2c[h];
  else if (q == 0)           v = c2p[h * 196 + (k - 1)];
  else if (k == 0)           v = p2c[h * 196 + (q - 1)];
  else {
    int p = q - 1, pk = k - 1;
    int dh = p / 14 - pk / 14 + 13;
    int dw = p % 14 - pk % 14 + 13;
    v = rel_table[(dh * 27 + dw) * NH + h];
  }
  biasF[(long)h * L_SEQ * L_SEQ + i] = v;
}

// ---------------------------------------------------------------- pad zeroing
__global__ void pad_kernel(_Float16* __restrict__ Qh, _Float16* __restrict__ Kh,
                           _Float16* __restrict__ Vt) {
  int i = blockIdx.x * 256 + threadIdx.x;
  const int NQ = NBH * 11 * HD;       // rows 197..207 of Q/K
  if (i < NQ) {
    int bh = i / (11 * HD); int rem = i - bh * 11 * HD;
    int l = 197 + rem / HD; int d = rem % HD;
    long off = ((long)bh * L_PAD + l) * HD + d;
    Qh[off] = (_Float16)0.f;
    Kh[off] = (_Float16)0.f;
  }
  const int NV = NBH * HD * 27;       // cols 197..223 of Vt
  if (i < NV) {
    int bh = i / (HD * 27); int rem = i - bh * HD * 27;
    int d = rem / 27; int l = 197 + rem % 27;
    Vt[((long)bh * HD + d) * L_VPAD + l] = (_Float16)0.f;
  }
}

// ---------------------------------------------------------------- WMMA GEMM
// Y = A(Nx768,f16) * W(768x768,f16) + bias. mode: 0 -> f16 (b,h,l,d) [Q/K],
// 2 -> f16 (b,h,d,l) padded [V^T], 3 -> fp32 row-major [final output].
// Tiles staged with double-buffered global_load_async_to_lds_b128 (ASYNCcnt).
__global__ __launch_bounds__(256) void gemm_proj_kernel(
    const _Float16* __restrict__ Ah, const _Float16* __restrict__ Wt,
    const float* __restrict__ bias, void* __restrict__ Out, int mode) {
  __shared__ __align__(16) _Float16 As[2][128][40];
  __shared__ __align__(16) _Float16 Bs[2][64][40];
  const int tid = threadIdx.x;
  const int lane = tid & 31, wid = tid >> 5;
  const int laneLo = lane & 15, laneHi = lane >> 4;
  const int wm = wid & 3, wn = wid >> 2;
  const int rowBase = blockIdx.x * 128;
  const int colBase = blockIdx.y * 64;

  v8f acc[2][2];
#pragma unroll
  for (int i = 0; i < 2; ++i)
#pragma unroll
    for (int j = 0; j < 2; ++j) acc[i][j] = v8f_zero();

  // 16B-chunk staging: A tile 128x32h (2 chunks/thread), B tile 64x32h (1/thread)
  const int r0 = tid >> 2, sg = tid & 3;
  auto stage = [&](int sbuf, int kk) {
    unsigned aBase = (unsigned)(uintptr_t)&As[sbuf][0][0];
    unsigned bBase = (unsigned)(uintptr_t)&Bs[sbuf][0][0];
    async_copy16(aBase + (unsigned)(r0 * 80 + sg * 16),
                 Ah + (long)(rowBase + r0) * CDIM + kk + sg * 8);
    async_copy16(aBase + (unsigned)((r0 + 64) * 80 + sg * 16),
                 Ah + (long)(rowBase + r0 + 64) * CDIM + kk + sg * 8);
    async_copy16(bBase + (unsigned)(r0 * 80 + sg * 16),
                 Wt + (long)(colBase + r0) * CDIM + kk + sg * 8);
  };

  stage(0, 0);
  int buf = 0;
  for (int k0 = 0; k0 < CDIM; k0 += 32) {
    if (k0 + 32 < CDIM) {
      stage(buf ^ 1, k0 + 32);
      asm volatile("s_wait_asynccnt 0x3" ::: "memory");  // current tile landed
    } else {
      asm volatile("s_wait_asynccnt 0x0" ::: "memory");
    }
    __syncthreads();

    F16Frag a0, a1, b0, b1;
#pragma unroll
    for (int j = 0; j < 8; ++j) {
      int kp = kpair(j, laneHi);
      a0.h[2 * j] = As[buf][wm * 32 + laneLo][kp];      a0.h[2 * j + 1] = As[buf][wm * 32 + laneLo][kp + 1];
      a1.h[2 * j] = As[buf][wm * 32 + 16 + laneLo][kp]; a1.h[2 * j + 1] = As[buf][wm * 32 + 16 + laneLo][kp + 1];
      b0.h[2 * j] = Bs[buf][wn * 32 + laneLo][kp];      b0.h[2 * j + 1] = Bs[buf][wn * 32 + laneLo][kp + 1];
      b1.h[2 * j] = Bs[buf][wn * 32 + 16 + laneLo][kp]; b1.h[2 * j + 1] = Bs[buf][wn * 32 + 16 + laneLo][kp + 1];
    }
    acc[0][0] = __builtin_amdgcn_wmma_f32_16x16x32_f16(false, a0.v, false, b0.v, (short)0, acc[0][0], false, false);
    acc[0][1] = __builtin_amdgcn_wmma_f32_16x16x32_f16(false, a0.v, false, b1.v, (short)0, acc[0][1], false, false);
    acc[1][0] = __builtin_amdgcn_wmma_f32_16x16x32_f16(false, a1.v, false, b0.v, (short)0, acc[1][0], false, false);
    acc[1][1] = __builtin_amdgcn_wmma_f32_16x16x32_f16(false, a1.v, false, b1.v, (short)0, acc[1][1], false, false);
    __syncthreads();
    buf ^= 1;
  }

#pragma unroll
  for (int fi = 0; fi < 2; ++fi)
#pragma unroll
    for (int fj = 0; fj < 2; ++fj)
#pragma unroll
      for (int r = 0; r < 8; ++r) {
        int row = rowBase + wm * 32 + fi * 16 + r + 8 * laneHi;
        int col = colBase + wn * 32 + fj * 16 + laneLo;
        float v = acc[fi][fj][r] + bias[col];
        if (row < NROWS) {
          if (mode == 3) {
            ((float*)Out)[(long)row * CDIM + col] = v;
          } else {
            int bb = row / L_SEQ;
            int l = row - bb * L_SEQ;
            int hh = col >> 6, d = col & 63;
            _Float16 hv = (_Float16)v;
            if (mode == 2)
              ((_Float16*)Out)[(((long)bb * NH + hh) * HD + d) * L_VPAD + l] = hv;
            else
              ((_Float16*)Out)[(((long)bb * NH + hh) * L_PAD + l) * HD + d] = hv;
          }
        }
      }
}

// ---------------------------------------------------------------- attention
__global__ __launch_bounds__(256) void attn_kernel(
    const _Float16* __restrict__ Qh, const _Float16* __restrict__ Kh,
    const _Float16* __restrict__ Vt, const float* __restrict__ biasF,
    _Float16* __restrict__ AO) {
  __shared__ _Float16 Ps[8][16][228];   // per-wave P strip, padded
  const int tid = threadIdx.x;
  const int lane = tid & 31, wid = tid >> 5;
  const int laneLo = lane & 15, laneHi = lane >> 4;
  const int bh = blockIdx.x;
  const int b = bh / NH, h = bh - b * NH;
  const _Float16* Qb = Qh + (long)bh * L_PAD * HD;
  const _Float16* Kb = Kh + (long)bh * L_PAD * HD;
  const _Float16* Vb = Vt + (long)bh * HD * L_VPAD;
  const float* biasH = biasF + (long)h * L_SEQ * L_SEQ;
  const float scale = 0.125f;

  for (int t = wid; t < 13; t += 8) {
    const int m0 = t * 16;
    // --- S = scale*(Q K^T) + bias : 13 N-tiles, K = 64 (2 wmma each) ---
    F16Frag aq0, aq1;
#pragma unroll
    for (int j = 0; j < 8; ++j) {
      int kp = kpair(j, laneHi);
      const _Float16* p0 = Qb + (m0 + laneLo) * HD + kp;
      aq0.h[2 * j] = p0[0];  aq0.h[2 * j + 1] = p0[1];
      aq1.h[2 * j] = p0[32]; aq1.h[2 * j + 1] = p0[33];
    }
    v8f acc[13];
#pragma unroll
    for (int n = 0; n < 13; ++n) acc[n] = v8f_zero();
#pragma unroll
    for (int n = 0; n < 13; ++n) {
      F16Frag bk0, bk1;
#pragma unroll
      for (int j = 0; j < 8; ++j) {
        int kp = kpair(j, laneHi);
        const _Float16* p0 = Kb + (n * 16 + laneLo) * HD + kp;
        bk0.h[2 * j] = p0[0];  bk0.h[2 * j + 1] = p0[1];
        bk1.h[2 * j] = p0[32]; bk1.h[2 * j + 1] = p0[33];
      }
      acc[n] = __builtin_amdgcn_wmma_f32_16x16x32_f16(false, aq0.v, false, bk0.v, (short)0, acc[n], false, false);
      acc[n] = __builtin_amdgcn_wmma_f32_16x16x32_f16(false, aq1.v, false, bk1.v, (short)0, acc[n], false, false);
    }
    // --- softmax over the 197 valid columns ---
    float rmax[8], rsum[8], rinv[8];
#pragma unroll
    for (int r = 0; r < 8; ++r) rmax[r] = -1e30f;
    const int mg_base = m0 + 8 * laneHi;
#pragma unroll
    for (int n = 0; n < 13; ++n) {
      int cg = n * 16 + laneLo;
#pragma unroll
      for (int r = 0; r < 8; ++r) {
        int mg = mg_base + r;
        float v = acc[n][r] * scale;
        if (cg < L_SEQ) {
          if (mg < L_SEQ) v += biasH[mg * L_SEQ + cg];
        } else {
          v = -1e30f;
        }
        acc[n][r] = v;
        rmax[r] = fmaxf(rmax[r], v);
      }
    }
#pragma unroll
    for (int m = 8; m >= 1; m >>= 1)
#pragma unroll
      for (int r = 0; r < 8; ++r)
        rmax[r] = fmaxf(rmax[r], __shfl_xor(rmax[r], m, 32));
#pragma unroll
    for (int r = 0; r < 8; ++r) rsum[r] = 0.f;
#pragma unroll
    for (int n = 0; n < 13; ++n)
#pragma unroll
      for (int r = 0; r < 8; ++r) {
        float p = __expf(acc[n][r] - rmax[r]);
        acc[n][r] = p;
        rsum[r] += p;
      }
#pragma unroll
    for (int m = 8; m >= 1; m >>= 1)
#pragma unroll
      for (int r = 0; r < 8; ++r)
        rsum[r] += __shfl_xor(rsum[r], m, 32);
#pragma unroll
    for (int r = 0; r < 8; ++r) rinv[r] = 1.0f / rsum[r];
    // --- write P (C-layout -> row-major LDS strip), zero pad cols ---
#pragma unroll
    for (int n = 0; n < 13; ++n)
#pragma unroll
      for (int r = 0; r < 8; ++r)
        Ps[wid][r + 8 * laneHi][n * 16 + laneLo] = (_Float16)(acc[n][r] * rinv[r]);
#pragma unroll
    for (int j = 0; j < 8; ++j)
      Ps[wid][laneLo][208 + 8 * laneHi + j] = (_Float16)0.f;
    // --- O = P V : K = 224 (7 steps), 4 N-tiles of 16 over d ---
    v8f vacc[4];
#pragma unroll
    for (int nd = 0; nd < 4; ++nd) vacc[nd] = v8f_zero();
    for (int kk = 0; kk < 7; ++kk) {
      F16Frag ap;
#pragma unroll
      for (int j = 0; j < 8; ++j) {
        int kp = kpair(j, laneHi) + kk * 32;
        ap.h[2 * j] = Ps[wid][laneLo][kp];
        ap.h[2 * j + 1] = Ps[wid][laneLo][kp + 1];
      }
#pragma unroll
      for (int nd = 0; nd < 4; ++nd) {
        F16Frag bv;
#pragma unroll
        for (int j = 0; j < 8; ++j) {
          int kp = kpair(j, laneHi) + kk * 32;
          const _Float16* p = Vb + (nd * 16 + laneLo) * L_VPAD + kp;
          bv.h[2 * j] = p[0]; bv.h[2 * j + 1] = p[1];
        }
        vacc[nd] = __builtin_amdgcn_wmma_f32_16x16x32_f16(false, ap.v, false, bv.v, (short)0, vacc[nd], false, false);
      }
    }
#pragma unroll
    for (int nd = 0; nd < 4; ++nd)
#pragma unroll
      for (int r = 0; r < 8; ++r) {
        int mg = mg_base + r;
        if (mg < L_SEQ) {
          int col = h * HD + nd * 16 + laneLo;
          AO[((long)b * L_SEQ + mg) * CDIM + col] = (_Float16)vacc[nd][r];
        }
      }
  }
}

// ---------------------------------------------------------------- launcher
extern "C" void kernel_launch(void* const* d_in, const int* in_sizes, int n_in,
                              void* d_out, int out_size, void* d_ws, size_t ws_size,
                              hipStream_t stream) {
  const float* X   = (const float*)d_in[0];
  const float* Wq  = (const float*)d_in[1];
  const float* bq  = (const float*)d_in[2];
  const float* Wk  = (const float*)d_in[3];
  const float* bk  = (const float*)d_in[4];
  const float* Wv  = (const float*)d_in[5];
  const float* bv  = (const float*)d_in[6];
  const float* Wo  = (const float*)d_in[7];
  const float* bo  = (const float*)d_in[8];
  const float* rel = (const float*)d_in[9];
  const float* c2p = (const float*)d_in[10];
  const float* p2c = (const float*)d_in[11];
  const float* c2c = (const float*)d_in[12];

  char* ws = (char*)d_ws;
  size_t off = 0;
  auto take = [&](size_t bytes) {
    char* p = ws + off;
    off += (bytes + 255) & ~(size_t)255;
    return (void*)p;
  };
  _Float16* WtQ = (_Float16*)take((size_t)CDIM * CDIM * 2);
  _Float16* WtK = (_Float16*)take((size_t)CDIM * CDIM * 2);
  _Float16* WtV = (_Float16*)take((size_t)CDIM * CDIM * 2);
  _Float16* WtO = (_Float16*)take((size_t)CDIM * CDIM * 2);
  float*    biasF = (float*)take((size_t)NH * L_SEQ * L_SEQ * 4);
  _Float16* Xh = (_Float16*)take((size_t)NROWS * CDIM * 2);
  _Float16* Qh = (_Float16*)take((size_t)NBH * L_PAD * HD * 2);
  _Float16* Kh = (_Float16*)take((size_t)NBH * L_PAD * HD * 2);
  _Float16* Vt = (_Float16*)take((size_t)NBH * HD * L_VPAD * 2);
  _Float16* AO = (_Float16*)take((size_t)NROWS * CDIM * 2);

  xprep_kernel<<<(NROWS * CDIM / 4) / 256, 256, 0, stream>>>(X, Xh);
  wprep_kernel<<<2304, 256, 0, stream>>>(Wq, WtQ);
  wprep_kernel<<<2304, 256, 0, stream>>>(Wk, WtK);
  wprep_kernel<<<2304, 256, 0, stream>>>(Wv, WtV);
  wprep_kernel<<<2304, 256, 0, stream>>>(Wo, WtO);
  bias_kernel<<<dim3((L_SEQ * L_SEQ + 255) / 256, NH), 256, 0, stream>>>(rel, c2p, p2c, c2c, biasF);
  pad_kernel<<<(NBH * HD * 27 + 255) / 256, 256, 0, stream>>>(Qh, Kh, Vt);

  dim3 ggrid(NROWS / 128, CDIM / 64);
  gemm_proj_kernel<<<ggrid, 256, 0, stream>>>(Xh, WtQ, bq, Qh, 0);
  gemm_proj_kernel<<<ggrid, 256, 0, stream>>>(Xh, WtK, bk, Kh, 0);
  gemm_proj_kernel<<<ggrid, 256, 0, stream>>>(Xh, WtV, bv, Vt, 2);

  attn_kernel<<<NBH, 256, 0, stream>>>(Qh, Kh, Vt, biasF, AO);

  gemm_proj_kernel<<<ggrid, 256, 0, stream>>>(AO, WtO, bo, d_out, 3);
}
